// AtomAttentionEncoder_77223511982613
// MI455X (gfx1250) — matrix operand order repaired and checked by reading, not compiled, not run on hardware
//
#include <hip/hip_runtime.h>
#include <hip/hip_bf16.h>

typedef __attribute__((ext_vector_type(16))) __bf16 v16bf;
typedef __attribute__((ext_vector_type(8)))  float  v8f;
typedef __attribute__((ext_vector_type(2)))  float  v2f;

#define N_ATOM 1536
#define N_TOKEN 384
#define NBR 32
#define C_ATOM 128
#define C_PAIR 32
#define C_TOKEN 384
#define NH 4
#define NL 3
#define DH 32
#define IN_ATOM 148
#define JTILES 96            // 1536/16 column tiles per atom row
#define TILES_PER_WAVE 8

#if __has_builtin(__builtin_amdgcn_global_load_async_to_lds_b32)
#define HAVE_ASYNC_LDS 1
typedef int __attribute__((address_space(1)))* gld_gptr;   // global (AS1) int*
typedef int __attribute__((address_space(3)))* gld_lptr;   // LDS (AS3) int*
#else
#define HAVE_ASYNC_LDS 0
#endif

// ---------------------------------------------------------------------------
// Stage 1: c_atom0 = concat(pos,charge,element,name)@W_atom + b + trunk_sing
// ---------------------------------------------------------------------------
__global__ void k_embed(const float* __restrict__ pos, const float* __restrict__ charge,
                        const float* __restrict__ elem, const float* __restrict__ namec,
                        const int* __restrict__ tok, const float* __restrict__ Wa,
                        const float* __restrict__ ba, const float* __restrict__ tsing,
                        float* __restrict__ c0_out, float* __restrict__ x_out) {
  __shared__ float sx[IN_ATOM];
  int i = blockIdx.x, t = threadIdx.x;
  if (t < 3)  sx[t] = pos[i * 3 + t];
  if (t == 3) sx[3] = charge[i];
  sx[4 + t] = elem[i * 128 + t];
  if (t < 16) sx[132 + t] = namec[i * 16 + t];
  __syncthreads();
  float acc = ba[t];
#pragma unroll 4
  for (int k = 0; k < IN_ATOM; ++k) acc = fmaf(sx[k], Wa[k * C_ATOM + t], acc);
  acc += tsing[tok[i] * C_ATOM + t];
  c0_out[i * C_ATOM + t] = acc;
  x_out[i * C_ATOM + t]  = acc;
}

// ---------------------------------------------------------------------------
// Stage 2: pair MLP  (bf16 WMMA 16x16x32, f32 accumulate) + trunk_pair add
// One wave = one (i, j0..j0+15) tile; 8 tiles per wave; 8 WMMAs per tile.
// ---------------------------------------------------------------------------
__global__ void k_pair(const float* __restrict__ pos, const int* __restrict__ tok,
                       const float* __restrict__ Wp, const float* __restrict__ bp,
                       const float* __restrict__ W1, const float* __restrict__ b1,
                       const float* __restrict__ W2, const float* __restrict__ b2,
                       const float* __restrict__ tpair, float* __restrict__ p_out) {
  __shared__ __bf16 hst[4][16][64];   // per-wave relu(h) staging (16 rows x 64)
  __shared__ int    stj[4][16];       // per-wave token id of the 16 j rows
  __shared__ float  swp[128], sbp[32];
  const int t = threadIdx.x, w = t >> 5, lane = t & 31, hi = lane >> 4, mn = lane & 15;
  if (t < 128) swp[t] = Wp[t];
  if (t < 32)  sbp[t] = bp[t];
  __syncthreads();

  // B fragments: lane holds column n = mn; element e -> K = (e<8?e:e+8) + 8*hi
  v16bf B1f[4];
#pragma unroll
  for (int n = 0; n < 4; ++n)
#pragma unroll
    for (int e = 0; e < 16; ++e)
      B1f[n][e] = (__bf16)W1[((e < 8 ? e : e + 8) + hi * 8) * 64 + n * 16 + mn];
  v16bf B2f[2][2];
#pragma unroll
  for (int kc = 0; kc < 2; ++kc)
#pragma unroll
    for (int n = 0; n < 2; ++n)
#pragma unroll
      for (int e = 0; e < 16; ++e)
        B2f[kc][n][e] = (__bf16)W2[(kc * 32 + (e < 8 ? e : e + 8) + hi * 8) * 32 + n * 16 + mn];

  const int wave = blockIdx.x * 4 + w;
  const long tile0 = (long)wave * TILES_PER_WAVE;
  const v8f z = {0.f, 0.f, 0.f, 0.f, 0.f, 0.f, 0.f, 0.f};

  for (int it = 0; it < TILES_PER_WAVE; ++it) {
    __syncthreads();                       // protect hst/stj reuse across iterations
    const long tile = tile0 + it;
    const int i  = (int)(tile / JTILES);
    const int j0 = (int)(tile % JTILES) * 16;

    const float pix = pos[i * 3], piy = pos[i * 3 + 1], piz = pos[i * 3 + 2];
    const int ti = tok[i];
    __builtin_prefetch(&tpair[(long)ti * N_TOKEN * C_PAIR], 0, 1);
    const int j = j0 + mn;
    const float dx = pix - pos[j * 3], dy = piy - pos[j * 3 + 1], dz = piz - pos[j * 3 + 2];
    const int tj = tok[j];
    const float same = (ti == tj) ? 1.f : 0.f;
    if (hi == 0) stj[w][mn] = tj;

    // A fragment of p0 (16 rows x 32 chan): element e -> K = (e<8?e:e+8)+8*hi
    v16bf a0;
#pragma unroll
    for (int e = 0; e < 16; ++e) {
      const int c = (e < 8 ? e : e + 8) + hi * 8;
      float v = fmaf(dx, swp[c], fmaf(dy, swp[32 + c],
                fmaf(dz, swp[64 + c], fmaf(same, swp[96 + c], sbp[c]))));
      a0[e] = (__bf16)v;
    }

    // mlp1: 4 N-tiles of 16, relu, restage as bf16
#pragma unroll
    for (int n = 0; n < 4; ++n) {
      v8f h = __builtin_amdgcn_wmma_f32_16x16x32_bf16(false, a0, false, B1f[n],
                                                      (short)0, z, false, false);
      const float bn = b1[n * 16 + mn];
#pragma unroll
      for (int r = 0; r < 8; ++r) {
        float v = h[r] + bn;
        v = v > 0.f ? v : 0.f;
        hst[w][r + hi * 8][n * 16 + mn] = (__bf16)v;
      }
    }
    __syncthreads();

    // re-form A fragments for mlp2 (K=64 -> two K=32 chunks)
    v16bf a2a, a2b;
#pragma unroll
    for (int e = 0; e < 16; ++e) {
      const int kl = (e < 8 ? e : e + 8) + hi * 8;
      a2a[e] = hst[w][mn][kl];
      a2b[e] = hst[w][mn][32 + kl];
    }
#pragma unroll
    for (int n = 0; n < 2; ++n) {
      v8f pa = __builtin_amdgcn_wmma_f32_16x16x32_bf16(false, a2a, false, B2f[0][n],
                                                       (short)0, z, false, false);
      pa = __builtin_amdgcn_wmma_f32_16x16x32_bf16(false, a2b, false, B2f[1][n],
                                                   (short)0, pa, false, false);
      const int c = n * 16 + mn;
      const float bc = b2[c];
#pragma unroll
      for (int r = 0; r < 8; ++r) {
        const int m = r + hi * 8;
        const int jr = j0 + m;
        float v = pa[r] + bc + tpair[((long)ti * N_TOKEN + stj[w][m]) * C_PAIR + c];
        p_out[((long)i * N_ATOM + jr) * C_PAIR + c] = v;
      }
    }
  }
}

// ---------------------------------------------------------------------------
// Generic f32 GEMM via V_WMMA_F32_16X16X4_F32: D = A(MxK) @ B(KxN) [+bias][+add]
// one wave per 16x16 output tile, K stepped by 4.
// ---------------------------------------------------------------------------
__global__ void k_gemm_f32(const float* __restrict__ A, const float* __restrict__ B,
                           const float* __restrict__ bias, const float* __restrict__ add,
                           float* __restrict__ D, int M, int N, int K) {
  const int t = threadIdx.x, w = t >> 5, lane = t & 31, hi = lane >> 4, mn = lane & 15;
  const int tid = blockIdx.x * 4 + w;
  const int ntile = N >> 4;
  const int tm = tid / ntile, tn = tid % ntile;
  const int row = tm * 16 + mn, col = tn * 16 + mn;
  v8f acc = {0.f, 0.f, 0.f, 0.f, 0.f, 0.f, 0.f, 0.f};
  for (int kk = 0; kk < K; kk += 4) {
    const int ka = kk + hi * 2;          // A: K = e + 2*hi ; B: same striping
    v2f a, b;
    a[0] = A[row * K + ka];
    a[1] = A[row * K + ka + 1];
    b[0] = B[ka * N + col];
    b[1] = B[(ka + 1) * N + col];
    acc = __builtin_amdgcn_wmma_f32_16x16x4_f32(false, a, false, b, (short)0, acc,
                                                false, false);
  }
#pragma unroll
  for (int r = 0; r < 8; ++r) {
    const int m = tm * 16 + r + hi * 8, n = tn * 16 + mn;
    float v = acc[r];
    if (bias) v += bias[n];
    if (add)  v += add[m * N + n];
    D[m * N + n] = v;
  }
}

// ---------------------------------------------------------------------------
// Per-atom neighborhood attention (one block per atom, 128 threads).
// Gathered k/v rows are staged to LDS with GLOBAL_LOAD_ASYNC_TO_LDS (ASYNCcnt).
// ---------------------------------------------------------------------------
__global__ void k_attn(const float* __restrict__ q, const float* __restrict__ k,
                       const float* __restrict__ v, const float* __restrict__ p,
                       const int* __restrict__ bidx, const float* __restrict__ Wb,
                       float* __restrict__ out) {
  __shared__ float s_k[NBR][C_ATOM];
  __shared__ float s_v[NBR][C_ATOM];
  __shared__ float s_sc[NBR][NH];
  __shared__ float s_q[C_ATOM];
  __shared__ int   s_i[NBR];
  const int i = blockIdx.x, t = threadIdx.x;
  if (t < NBR) s_i[t] = bidx[i * NBR + t];
  s_q[t] = q[i * C_ATOM + t];
  __syncthreads();

#if HAVE_ASYNC_LDS
#pragma unroll 4
  for (int b = 0; b < NBR; ++b) {
    const int rowk = s_i[b] * C_ATOM + t;
    __builtin_amdgcn_global_load_async_to_lds_b32(
        (gld_gptr)(void*)(k + rowk), (gld_lptr)(void*)&s_k[b][t], 0, 0);
    __builtin_amdgcn_global_load_async_to_lds_b32(
        (gld_gptr)(void*)(v + rowk), (gld_lptr)(void*)&s_v[b][t], 0, 0);
  }
  asm volatile("s_wait_asynccnt 0x0" ::: "memory");
#else
#pragma unroll 4
  for (int b = 0; b < NBR; ++b) {
    const int rowk = s_i[b] * C_ATOM + t;
    s_k[b][t] = k[rowk];
    s_v[b][t] = v[rowk];
  }
#endif
  __syncthreads();

  const int b = t & 31, h = t >> 5;
  const float* prow = p + ((long)i * N_ATOM + s_i[b]) * C_PAIR;
  float bias = 0.f;
#pragma unroll 8
  for (int c = 0; c < C_PAIR; ++c) bias = fmaf(prow[c], Wb[c * NH + h], bias);
  float sc = 0.f;
#pragma unroll 8
  for (int d = 0; d < DH; ++d) sc = fmaf(s_q[h * DH + d], s_k[b][h * DH + d], sc);
  s_sc[b][h] = sc * 0.17677669529663687f + bias;
  __syncthreads();

  const int d = t & 31;                  // reuse mapping: (h, d)
  float mx = -1e30f;
#pragma unroll
  for (int bb = 0; bb < NBR; ++bb) mx = fmaxf(mx, s_sc[bb][h]);
  float den = 0.f;
#pragma unroll
  for (int bb = 0; bb < NBR; ++bb) den += __expf(s_sc[bb][h] - mx);
  const float inv = 1.f / den;
  float o = 0.f;
  for (int bb = 0; bb < NBR; ++bb) {
    const float a = __expf(s_sc[bb][h] - mx) * inv;
    o = fmaf(a, s_v[bb][h * DH + d], o);
  }
  out[i * C_ATOM + h * DH + d] = o;
}

__global__ void k_copy(const float* __restrict__ src, float* __restrict__ dst, int n) {
  int i = blockIdx.x * blockDim.x + threadIdx.x;
  if (i < n) dst[i] = src[i];
}

// sorted-segment mean: one block per token, one thread per channel
__global__ void k_segmean(const float* __restrict__ qp, const int* __restrict__ tok,
                          float* __restrict__ out) {
  __shared__ int s_lo, s_hi;
  const int tkn = blockIdx.x, c = threadIdx.x;
  if (c == 0) {
    int a = 0, b = N_ATOM;
    while (a < b) { int m = (a + b) >> 1; if (tok[m] < tkn) a = m + 1; else b = m; }
    s_lo = a;
    b = N_ATOM;
    while (a < b) { int m = (a + b) >> 1; if (tok[m] < tkn + 1) a = m + 1; else b = m; }
    s_hi = a;
  }
  __syncthreads();
  const int lo = s_lo, hiE = s_hi, cnt = hiE - lo;
  float s = 0.f;
  for (int a = lo; a < hiE; ++a) s += qp[a * C_TOKEN + c];
  out[tkn * C_TOKEN + c] = s / (float)(cnt > 0 ? cnt : 1);
}

// ---------------------------------------------------------------------------
extern "C" void kernel_launch(void* const* d_in, const int* in_sizes, int n_in,
                              void* d_out, int out_size, void* d_ws, size_t ws_size,
                              hipStream_t stream) {
  const float* ref_pos    = (const float*)d_in[0];
  const float* ref_charge = (const float*)d_in[1];
  const float* ref_elem   = (const float*)d_in[2];
  const float* ref_name   = (const float*)d_in[3];
  const int*   a2t        = (const int*)d_in[4];
  /* d_in[5] restype: unused by reference math */
  const float* trunk_sing = (const float*)d_in[6];
  const float* trunk_pair = (const float*)d_in[7];
  const int*   bidx       = (const int*)d_in[8];
  const float* W_atom = (const float*)d_in[9];
  const float* b_atom = (const float*)d_in[10];
  const float* W_pair = (const float*)d_in[11];
  const float* b_pair = (const float*)d_in[12];
  const float* W1     = (const float*)d_in[13];
  const float* b1     = (const float*)d_in[14];
  const float* W2     = (const float*)d_in[15];
  const float* b2     = (const float*)d_in[16];
  const float* Wq     = (const float*)d_in[17];
  const float* Wk     = (const float*)d_in[18];
  const float* Wv     = (const float*)d_in[19];
  const float* Wo     = (const float*)d_in[20];
  const float* Wb     = (const float*)d_in[21];
  const float* W_post = (const float*)d_in[22];
  const float* b_post = (const float*)d_in[23];

  float* out        = (float*)d_out;
  float* a_token    = out;                              // 384*384
  float* q_atom_out = out + N_TOKEN * C_TOKEN;          // 1536*128
  float* c0_out     = q_atom_out + N_ATOM * C_ATOM;     // 1536*128
  float* p_out      = c0_out + N_ATOM * C_ATOM;         // 1536*1536*32

  float* ws = (float*)d_ws;
  float* xa    = ws;                    // 196608
  float* xb    = xa + N_ATOM * C_ATOM;
  float* qb    = xb + N_ATOM * C_ATOM;
  float* kb    = qb + N_ATOM * C_ATOM;
  float* vb    = kb + N_ATOM * C_ATOM;
  float* ob    = vb + N_ATOM * C_ATOM;
  float* qproj = ob + N_ATOM * C_ATOM;  // 1536*384

  k_embed<<<N_ATOM, 128, 0, stream>>>(ref_pos, ref_charge, ref_elem, ref_name, a2t,
                                      W_atom, b_atom, trunk_sing, c0_out, xa);

  // 1536*96 tiles / (8 tiles/wave) / (4 waves/block) = 4608 blocks
  k_pair<<<4608, 128, 0, stream>>>(ref_pos, a2t, W_pair, b_pair, W1, b1, W2, b2,
                                   trunk_pair, p_out);

  float* xc = xa;
  float* xn = xb;
  for (int l = 0; l < NL; ++l) {
    k_gemm_f32<<<192, 128, 0, stream>>>(xc, Wq + l * C_ATOM * C_ATOM, nullptr, nullptr,
                                        qb, N_ATOM, C_ATOM, C_ATOM);
    k_gemm_f32<<<192, 128, 0, stream>>>(xc, Wk + l * C_ATOM * C_ATOM, nullptr, nullptr,
                                        kb, N_ATOM, C_ATOM, C_ATOM);
    k_gemm_f32<<<192, 128, 0, stream>>>(xc, Wv + l * C_ATOM * C_ATOM, nullptr, nullptr,
                                        vb, N_ATOM, C_ATOM, C_ATOM);
    k_attn<<<N_ATOM, 128, 0, stream>>>(qb, kb, vb, p_out, bidx, Wb + l * C_PAIR * NH, ob);
    k_gemm_f32<<<192, 128, 0, stream>>>(ob, Wo + l * C_ATOM * C_ATOM, nullptr, xc, xn,
                                        N_ATOM, C_ATOM, C_ATOM);
    float* tmp = xc; xc = xn; xn = tmp;
  }

  k_copy<<<768, 256, 0, stream>>>(xc, q_atom_out, N_ATOM * C_ATOM);
  k_gemm_f32<<<576, 128, 0, stream>>>(xc, W_post, b_post, nullptr, qproj,
                                      N_ATOM, C_TOKEN, C_ATOM);
  k_segmean<<<N_TOKEN, C_TOKEN, 0, stream>>>(qproj, a2t, a_token);
}
